// VanillaRNN_52707838656553
// MI455X (gfx1250) — compile-verified, matrix operand order checked by reading, not run on hardware
//
#include <hip/hip_runtime.h>
#include <hip/hip_bf16.h>

// ---------------------------------------------------------------------------
// CDNA5 (gfx1250) WMMA implementation of a 2-layer vanilla RNN LM.
//   V=32000, E=H=1024, B=16, S=256
// GEMMs: v_wmma_f32_16x16x32_bf16 (bf16 in, f32 accum)
// Staging: global_load_async_to_lds_b128 (immediate offsets, double-buffered)
// ---------------------------------------------------------------------------

typedef __attribute__((ext_vector_type(16))) __bf16 v16bf;
typedef __attribute__((ext_vector_type(8)))  float  v8f;
typedef __attribute__((ext_vector_type(4)))  __bf16 v4bf;

union FragU { v16bf v; uint4 q[2]; };

// Load a 16x32 bf16 WMMA operand fragment from a row-major [rows x ldk] array.
// Works for A (MxK) and for B supplied as N x K (i.e. W row-major, computing X*W^T).
// Per ISA 7.12.2 (16-bit A 16x32): lane r (=lane&15), half hs (=lane>>4):
//   VGPRs 0-3 hold K = hs*8 .. hs*8+7 ; VGPRs 4-7 hold K = 16+hs*8 .. 16+hs*8+7
__device__ __forceinline__ v16bf load_frag(const __bf16* base, int row0, int ldk,
                                           int kc, int lane) {
  const int r = lane & 15, hs = lane >> 4;
  const __bf16* p = base + (size_t)(row0 + r) * ldk + kc + hs * 8;
  FragU f;
  f.q[0] = *(const uint4*)(p);        // K segment 0 (8 halves, 16B aligned)
  f.q[1] = *(const uint4*)(p + 16);   // K segment +16
  return f.v;
}

__device__ __forceinline__ v8f wmma_bf16(v16bf a, v16bf b, v8f c) {
  // (neg_a, A, neg_b, B, c_mod, C, reuse_a, reuse_b)
  return __builtin_amdgcn_wmma_f32_16x16x32_bf16(false, a, false, b,
                                                 (short)0, c, false, false);
}

// --- CDNA5 async global->LDS (GV mode) --------------------------------------
// ISA 15.18: the immediate offset is added to both the LDS and the global address
// -> one immediate offset advances BOTH sides; share base VGPRs across copies.
template <int OFF>
__device__ __forceinline__ void async_b128o(unsigned lds_off, const void* g) {
  asm volatile("global_load_async_to_lds_b128 %0, %1, off offset:%2"
               :: "v"(lds_off), "v"(g), "i"(OFF) : "memory");
}
template <int N>
__device__ __forceinline__ void s_wait_async() {
  asm volatile("s_wait_asynccnt %0" :: "i"(N) : "memory");
}
__device__ __forceinline__ unsigned lds_off_of(const void* p) {
  // generic pointer to LDS: addr[31:0] is the wave-relative LDS byte offset
  return (unsigned)(uintptr_t)p;
}

// ---------------------------------------------------------------------------
// f32 -> bf16 conversion (vectorized x4, grid-stride)
// ---------------------------------------------------------------------------
__global__ void cvt_f32_bf16_v4(const float* __restrict__ src,
                                __bf16* __restrict__ dst, long n4) {
  long i = (long)blockIdx.x * blockDim.x + threadIdx.x;
  long stride = (long)gridDim.x * blockDim.x;
  for (; i < n4; i += stride) {
    float4 v = ((const float4*)src)[i];
    v4bf o = { (__bf16)v.x, (__bf16)v.y, (__bf16)v.z, (__bf16)v.w };
    ((v4bf*)dst)[i] = o;
  }
}

// ---------------------------------------------------------------------------
// Embedding gather + f32->bf16: xe[bs][e] = bf16(emb[x[bs]][e])
// grid = B*S blocks, 256 threads, 4 elems/thread (E=1024)
// ---------------------------------------------------------------------------
__global__ void gather_embed(const int* __restrict__ x,
                             const float* __restrict__ emb,
                             __bf16* __restrict__ xe) {
  const int bs = blockIdx.x;
  const int tok = x[bs];
  const int e = threadIdx.x * 4;
  float4 v = *(const float4*)(emb + (size_t)tok * 1024 + e);
  v4bf o = { (__bf16)v.x, (__bf16)v.y, (__bf16)v.z, (__bf16)v.w };
  *(v4bf*)(xe + (size_t)bs * 1024 + e) = o;
}

// ---------------------------------------------------------------------------
// Tiled WMMA GEMM:  C[M,N] = A[M,K] * B[N,K]^T + bias0[N] (+ bias1[N])
// Block tile 128x128, K-tile 64. 256 threads = 8 waves (2 M-groups x 4 N-groups),
// wave tile 64x32 = 4x2 WMMA accumulators (8 WMMAs / 12 ds_load_b128 per chunk).
// Double-buffered async global->LDS staging (8 async b128 per thread per tile).
// ---------------------------------------------------------------------------
__global__ __launch_bounds__(256) void gemm_bf16_bias(
    const __bf16* __restrict__ A, const __bf16* __restrict__ B,
    float* __restrict__ C, const float* __restrict__ bias0,
    const float* __restrict__ bias1, int M, int N, int K) {
  constexpr int TM = 128, TN = 128, TK = 64, LDS_K = 72;  // 72: bank pad, 16B-mult
  __shared__ __bf16 sA[2][TM * LDS_K];
  __shared__ __bf16 sB[2][TN * LDS_K];

  const int tid = threadIdx.x, lane = tid & 31, wv = tid >> 5;
  const int m0 = blockIdx.y * TM, n0 = blockIdx.x * TN;
  const int wm = (wv >> 2) * 64;  // wave M offset in block tile (2 groups)
  const int wn = (wv & 3) * 32;   // wave N offset in block tile (4 groups)

  // staging: 2 threads/row, 32 halves (64B) each, 4 async b128 per matrix
  const int row = tid >> 1, half = tid & 1;
  const __bf16* aSrcBase = A + (size_t)(m0 + row) * K + half * 32;
  const __bf16* bSrcBase = B + (size_t)(n0 + row) * K + half * 32;
  unsigned aDst[2], bDst[2];
  aDst[0] = lds_off_of(&sA[0][row * LDS_K + half * 32]);
  aDst[1] = lds_off_of(&sA[1][row * LDS_K + half * 32]);
  bDst[0] = lds_off_of(&sB[0][row * LDS_K + half * 32]);
  bDst[1] = lds_off_of(&sB[1][row * LDS_K + half * 32]);

  auto issue_tile = [&](int k0, int buf) {
    const __bf16* as = aSrcBase + k0;
    async_b128o<0>(aDst[buf], as);
    async_b128o<16>(aDst[buf], as);
    async_b128o<32>(aDst[buf], as);
    async_b128o<48>(aDst[buf], as);
    const __bf16* bs = bSrcBase + k0;
    async_b128o<0>(bDst[buf], bs);
    async_b128o<16>(bDst[buf], bs);
    async_b128o<32>(bDst[buf], bs);
    async_b128o<48>(bDst[buf], bs);
  };

  v8f acc[4][2] = {};

  issue_tile(0, 0);
  int buf = 0;
  for (int k0 = 0; k0 < K; k0 += TK, buf ^= 1) {
    if (k0 + TK < K) {
      issue_tile(k0 + TK, buf ^ 1);
      s_wait_async<8>();   // async ops complete in order: tile k0's 8 done
    } else {
      s_wait_async<0>();
    }
    __syncthreads();       // all waves' tile data visible in LDS

    const __bf16* cA = sA[buf];
    const __bf16* cB = sB[buf];
#pragma unroll
    for (int kc = 0; kc < TK; kc += 32) {
      v16bf b0 = load_frag(cB, wn,      LDS_K, kc, lane);
      v16bf b1 = load_frag(cB, wn + 16, LDS_K, kc, lane);
#pragma unroll
      for (int i = 0; i < 4; ++i) {
        v16bf a = load_frag(cA, wm + i * 16, LDS_K, kc, lane);
        acc[i][0] = wmma_bf16(a, b0, acc[i][0]);
        acc[i][1] = wmma_bf16(a, b1, acc[i][1]);
      }
    }
    __syncthreads();       // reads done before next-next tile overwrites buf
  }

  // C layout (ISA 7.12.2): lane = column (r), VGPR e = row hs*8+e
  const int r = lane & 15, hs = lane >> 4;
#pragma unroll
  for (int i = 0; i < 4; ++i)
#pragma unroll
    for (int j = 0; j < 2; ++j) {
      const int mb = m0 + wm + i * 16 + hs * 8;
      const int nc = n0 + wn + j * 16 + r;
      const float bv = (bias0 ? bias0[nc] : 0.f) + (bias1 ? bias1[nc] : 0.f);
#pragma unroll
      for (int e = 0; e < 8; ++e)
        C[(size_t)(mb + e) * N + nc] = acc[i][j][e] + bv;
    }
}

// ---------------------------------------------------------------------------
// Recurrent scan for one RNN layer:
//   h_t = tanh(pre[:, t, :] + h_{t-1} @ Whh^T)        (B=16, H=1024, S=256)
// Single 1024-thread workgroup (32 waves). h kept in LDS as bf16 [16][1032].
// Wave w owns output columns [32w, 32w+32): two 16x16 WMMA tiles per step.
// Whh is bf16 [H][H] row-major (streams from L2, ~2MB, reused 256x).
// ---------------------------------------------------------------------------
__global__ __launch_bounds__(1024) void rnn_scan(
    const float* __restrict__ pre, const __bf16* __restrict__ Whh,
    __bf16* __restrict__ out, float* __restrict__ hT) {
  constexpr int H = 1024, S = 256, LDH = 1032;  // padded stride (16B-mult)
  __shared__ __bf16 h[16 * LDH];

  const int tid = threadIdx.x, lane = tid & 31, wv = tid >> 5;
  const int n0 = wv * 32;
  const int r = lane & 15, hs = lane >> 4;

  for (int i = tid; i < 16 * LDH; i += 1024) h[i] = (__bf16)0.f;
  __syncthreads();

  for (int t = 0; t < S; ++t) {
    v8f acc[2];
#pragma unroll
    for (int j = 0; j < 2; ++j) {
      const int nc = n0 + j * 16 + r;
#pragma unroll
      for (int e = 0; e < 8; ++e)
        acc[j][e] = pre[(size_t)((hs * 8 + e) * S + t) * H + nc];
    }
    for (int kc = 0; kc < H; kc += 32) {
      v16bf a = load_frag(h, 0, LDH, kc, lane);  // h is 16 rows (batch) x H
#pragma unroll
      for (int j = 0; j < 2; ++j) {
        v16bf b = load_frag(Whh, n0 + j * 16, H, kc, lane);
        acc[j] = wmma_bf16(a, b, acc[j]);
      }
    }
    __syncthreads();  // all reads of h done before overwrite
#pragma unroll
    for (int j = 0; j < 2; ++j) {
      const int nc = n0 + j * 16 + r;
#pragma unroll
      for (int e = 0; e < 8; ++e) {
        const int b = hs * 8 + e;
        const __bf16 bv = (__bf16)tanhf(acc[j][e]);
        h[b * LDH + nc] = bv;
        out[(size_t)(b * S + t) * H + nc] = bv;
      }
    }
    __syncthreads();  // h_t visible before next step's reads
  }

  // final hidden state (f32)
#pragma unroll
  for (int j = 0; j < 2; ++j) {
    const int nc = n0 + j * 16 + r;
#pragma unroll
    for (int e = 0; e < 8; ++e) {
      const int b = hs * 8 + e;
      hT[b * H + nc] = (float)h[b * LDH + nc];
    }
  }
}

// ---------------------------------------------------------------------------
extern "C" void kernel_launch(void* const* d_in, const int* in_sizes, int n_in,
                              void* d_out, int out_size, void* d_ws, size_t ws_size,
                              hipStream_t stream) {
  (void)in_sizes; (void)n_in; (void)out_size; (void)ws_size;
  const int*   x    = (const int*)d_in[0];
  const float* emb  = (const float*)d_in[1];
  const float* Wih0 = (const float*)d_in[2];
  const float* Whh0 = (const float*)d_in[3];
  const float* bih0 = (const float*)d_in[4];
  const float* bhh0 = (const float*)d_in[5];
  const float* Wih1 = (const float*)d_in[6];
  const float* Whh1 = (const float*)d_in[7];
  const float* bih1 = (const float*)d_in[8];
  const float* bhh1 = (const float*)d_in[9];
  const float* fcW  = (const float*)d_in[10];
  const float* fcb  = (const float*)d_in[11];

  constexpr int B = 16, S = 256, E = 1024, H = 1024, V = 32000;
  constexpr int BS = B * S;  // 4096

  char* ws = (char*)d_ws;
  size_t off = 0;
  auto alloc = [&](size_t bytes) -> char* {
    char* p = ws + off;
    off += (bytes + 255) & ~(size_t)255;
    return p;
  };
  __bf16* xe   = (__bf16*)alloc((size_t)BS * E * 2);  // reused as out0 after pre0
  __bf16* o1   = (__bf16*)alloc((size_t)BS * H * 2);  // layer1 output (bf16)
  float*  pre  = (float*) alloc((size_t)BS * H * 4);  // reused for pre0/pre1
  __bf16* wih0 = (__bf16*)alloc((size_t)H * E * 2);
  __bf16* whh0 = (__bf16*)alloc((size_t)H * H * 2);
  __bf16* wih1 = (__bf16*)alloc((size_t)H * H * 2);
  __bf16* whh1 = (__bf16*)alloc((size_t)H * H * 2);
  __bf16* fcw  = (__bf16*)alloc((size_t)V * H * 2);

  float* logits = (float*)d_out;                       // [B,S,V]
  float* hid    = logits + (size_t)BS * V;             // [2,B,H]

  // 1) weights -> bf16
  cvt_f32_bf16_v4<<<1024, 256, 0, stream>>>(Wih0, wih0, (long)H * E / 4);
  cvt_f32_bf16_v4<<<1024, 256, 0, stream>>>(Whh0, whh0, (long)H * H / 4);
  cvt_f32_bf16_v4<<<1024, 256, 0, stream>>>(Wih1, wih1, (long)H * H / 4);
  cvt_f32_bf16_v4<<<1024, 256, 0, stream>>>(Whh1, whh1, (long)H * H / 4);
  cvt_f32_bf16_v4<<<4096, 256, 0, stream>>>(fcW,  fcw,  (long)V * H / 4);

  // 2) embedding gather (+convert)
  gather_embed<<<BS, 256, 0, stream>>>(x, emb, xe);

  // 3) pre0 = xe @ Wih0^T + b_ih0 + b_hh0
  gemm_bf16_bias<<<dim3(H / 128, BS / 128), 256, 0, stream>>>(
      xe, wih0, pre, bih0, bhh0, BS, H, E);

  // 4) layer-0 scan -> out0 (into xe buffer), hT0
  rnn_scan<<<1, 1024, 0, stream>>>(pre, whh0, xe, hid);

  // 5) pre1 = out0 @ Wih1^T + b_ih1 + b_hh1
  gemm_bf16_bias<<<dim3(H / 128, BS / 128), 256, 0, stream>>>(
      xe, wih1, pre, bih1, bhh1, BS, H, H);

  // 6) layer-1 scan -> out1, hT1
  rnn_scan<<<1, 1024, 0, stream>>>(pre, whh1, o1, hid + (size_t)B * H);

  // 7) logits = out1 @ fcW^T + fcb   (the 268-GFLOP GEMM)
  gemm_bf16_bias<<<dim3(V / 128, BS / 128), 256, 0, stream>>>(
      o1, fcw, logits, fcb, nullptr, BS, V, H);
}
